// TextEmbedding2_35613868818659
// MI455X (gfx1250) — matrix operand order repaired and checked by reading, not compiled, not run on hardware
//
#include <hip/hip_runtime.h>

typedef __attribute__((ext_vector_type(16))) _Float16 v16h;
typedef __attribute__((ext_vector_type(8)))  float    v8f;

#define SEQ   2048
#define NUMA  32
#define DIM   512
#define NK    96            // 3 * 32 virtual annotation rows
#define TT    128           // t-rows per workgroup (8 waves * 16)
#define ETPAD 112           // padded K-extent of Et rows (halves); 224B = 32B-aligned rows

__global__ void __launch_bounds__(256)
text_embed_wmma_kernel(const int* __restrict__ starts,
                       const int* __restrict__ ends,
                       const int* __restrict__ labels,
                       const float* __restrict__ emb_ing,
                       const float* __restrict__ emb_start,
                       const float* __restrict__ emb_end,
                       float* __restrict__ out)
{
    extern __shared__ char smem[];
    int* sh_s = (int*)smem;               // 32 clamped starts
    int* sh_e = sh_s + NUMA;              // 32 clamped ends
    int* sh_l = sh_e + NUMA;              // 32 labels
    _Float16* Et = (_Float16*)(smem + 512);   // [DIM][ETPAD] f16, transposed E

    const int tid  = threadIdx.x;
    const int b    = blockIdx.x >> 4;     // 16 t-tiles per batch
    const int tile = blockIdx.x & 15;

    // ---- load + clamp annotation data for this batch ----
    if (tid < NUMA) {
        int s = starts[b * NUMA + tid];
        int e = ends  [b * NUMA + tid];
        s = s < 0 ? 0 : (s > SEQ - 1 ? SEQ - 1 : s);
        e = e < 0 ? 0 : (e > SEQ - 1 ? SEQ - 1 : e);
        sh_s[tid] = s;
        sh_e[tid] = e;
        sh_l[tid] = labels[b * NUMA + tid];
    }
    __syncthreads();

    // ---- build Et[d][k] = valid ? table(type)[label][d] : 0   (f16, transposed) ----
    for (int idx = tid; idx < NK * DIM; idx += 256) {
        int d    = idx & (DIM - 1);
        int k    = idx >> 9;              // 0..95
        int a    = k & (NUMA - 1);
        int type = k >> 5;                // 0=ing, 1=start, 2=end
        const float* tbl = (type == 0) ? emb_ing : (type == 1) ? emb_start : emb_end;
        float v = (sh_s[a] < sh_e[a]) ? tbl[sh_l[a] * DIM + d] : 0.0f;
        Et[d * ETPAD + k] = (_Float16)v;
    }
    __syncthreads();

    // ---- WMMA: out_tile[16t x 16d] = Mask[16 x 96] @ E[96 x 16] ----
    const int lane = tid & 31;
    const int wave = tid >> 5;            // 8 waves -> 8 t-blocks of 16
    const int g    = lane >> 4;           // lane group (ISA f16 A/B layouts)
    const int m    = lane & 15;
    const int t0   = tile * TT + wave * 16;
    const int t    = t0 + m;              // A-matrix row (t index)

    // A fragments: f16 16x32 layout -> element i of v16h holds K = klocal(i)
    //   K = 16*g + i
    v16h A[3];
#pragma unroll
    for (int i = 0; i < 16; ++i) {
        int a = 16 * g + i;
        int s = sh_s[a];
        int e = sh_e[a];
        A[0][i] = (s <= t && t < e) ? (_Float16)1.0f : (_Float16)0.0f; // interval
        A[1][i] = (t == s)          ? (_Float16)1.0f : (_Float16)0.0f; // start point
        A[2][i] = (t == e)          ? (_Float16)1.0f : (_Float16)0.0f; // end point
    }

    float* ob = out + ((size_t)b * SEQ + t0) * DIM + m;

    for (int n0 = 0; n0 < DIM / 16; ++n0) {
        v8f acc = {};
        // B fragment: 32x16 f16 layout -> lane holds N = m, element i holds K = 32c + 16g + i
        const _Float16* ecol = Et + (n0 * 16 + m) * ETPAD + 16 * g;
#pragma unroll
        for (int c = 0; c < 3; ++c) {
            v16h B = *(const v16h*)(ecol + 32 * c);   // 32B contiguous, 32B-aligned LDS read
            acc = __builtin_amdgcn_wmma_f32_16x16x32_f16(
                false, A[c], false, B, (short)0, acc, false, false);
        }
        // C layout: VGPR r -> M = 8g + r, N = m ; one coalesced row-segment per store
        float* op = ob + n0 * 16;
#pragma unroll
        for (int r = 0; r < 8; ++r) {
            op[(size_t)(8 * g + r) * DIM] = acc[r];
        }
    }
}

extern "C" void kernel_launch(void* const* d_in, const int* in_sizes, int n_in,
                              void* d_out, int out_size, void* d_ws, size_t ws_size,
                              hipStream_t stream) {
    const int*   starts    = (const int*)  d_in[0];
    const int*   ends      = (const int*)  d_in[1];
    const int*   labels    = (const int*)  d_in[2];
    const float* emb_ing   = (const float*)d_in[3];
    const float* emb_start = (const float*)d_in[4];
    const float* emb_end   = (const float*)d_in[5];
    float* out = (float*)d_out;

    size_t shmem = 512 + (size_t)DIM * ETPAD * 2;   // indices + f16 Et
    dim3 grid(64 * (SEQ / TT));                     // 64 batches * 16 t-tiles = 1024
    dim3 block(256);                                // 8 wave32
    text_embed_wmma_kernel<<<grid, block, shmem, stream>>>(
        starts, ends, labels, emb_ing, emb_start, emb_end, out);
}
